// MultiHeadTargetAttention_CollectiveKV_51616916963440
// MI455X (gfx1250) — matrix operand order repaired and checked by reading, not compiled, not run on hardware
//
#include <hip/hip_runtime.h>

typedef __attribute__((ext_vector_type(16))) __bf16 v16bf;
typedef __attribute__((ext_vector_type(8)))  __bf16 v8bf;
typedef __attribute__((ext_vector_type(8)))  float  v8f;

#define EMB    128
#define MPOOL  10000
#define NTILES 625          // 10000 / 16
#define GD     120
#define ROWS   10240        // b*s = 256*40
#define NB     256          // batch

__device__ __forceinline__ float xor16(float x) {
  // ds_swizzle SWAPX16: group-of-32, xor=0x10, and=0x1f -> offset 0x401F
  return __int_as_float(__builtin_amdgcn_ds_swizzle(__float_as_int(x), 0x401F));
}

__device__ __forceinline__ float silu(float x) { return x / (1.f + __expf(-x)); }

__global__ void cast_hist_kernel(const float* __restrict__ in, __bf16* __restrict__ out, int n) {
  int t = blockIdx.x * 256 + threadIdx.x;
  if (t < n) out[t] = (__bf16)in[t];
}

// in: [128][10000] f32 row-major  ->  out: [10000][128] bf16 (N-major for WMMA B frags)
__global__ void transpose_cast_kernel(const float* __restrict__ in, __bf16* __restrict__ out) {
  int t = blockIdx.x * 256 + threadIdx.x;
  if (t >= MPOOL * EMB) return;
  int e = t & 127;
  int m = t >> 7;
  out[t] = (__bf16)in[(size_t)e * MPOOL + m];
}

// One WG per 16-row block. bf16 WMMA router GEMM fused with softmax statistics.
// Pass 1: per-slot running max + unstabilized sum(exp(x)) (branchless; logits are
// O(few), exp never overflows fp32). Pass 2: bit-identical WMMA recompute gives
// argmax via x>=rowmax, and accumulates exp(x)/rowsum into the global avg.
__global__ __launch_bounds__(256) void router_kernel(
    const __bf16* __restrict__ histb,    // [ROWS][128] bf16
    const __bf16* __restrict__ wt,       // [10000][128] bf16 (transposed router W)
    const float*  __restrict__ hist_f32, // [ROWS][128]
    const float*  __restrict__ Wu,       // [128][8]
    const float*  __restrict__ pool,     // [10000][120]
    float*        __restrict__ KV,       // [ROWS][128] out (silu applied)
    float*        __restrict__ avg_acc,  // [10000] prob-sum accumulator
    float*        __restrict__ lp_acc)   // [1] sum of -log(sigmoid(top))
{
  const int tid  = threadIdx.x;
  const int lane = tid & 31;
  const int wv   = tid >> 5;
  const int R0   = blockIdx.x * 16;
  const int hi   = (lane >= 16);
  const int l15  = lane & 15;

  __shared__ float red_m[256 * 8];
  __shared__ float red_s[256 * 8];
  __shared__ int   red_i[256 * 8];
  __shared__ float rowMax[16], rowInv[16], rowW[16];
  __shared__ int   rowIdx[16];

  union AB { v16bf v; v8bf h[2]; };

  // ---- A fragment: 16 rows x 128 K, resident in VGPRs for both passes ----
  AB a[4];
  {
    const __bf16* base = histb + (size_t)(R0 + l15) * EMB;
    const int off = hi ? 16 : 0;       // contiguous k order (k=16h+s, matched by B): lanes>=16 hold K=16..31
#pragma unroll
    for (int c = 0; c < 4; ++c) {
      a[c].h[0] = *(const v8bf*)(base + c * 32 + off);
      a[c].h[1] = *(const v8bf*)(base + c * 32 + 8 + off);
    }
  }

  const int boff = hi ? 16 : 0;        // contiguous k order, same as A: lanes>=16 hold K=16..31

  // ---- Pass 1: branchless running max + sum(exp(x)) per (row,lane) slot ----
  float m[8], s[8];
#pragma unroll
  for (int v = 0; v < 8; ++v) { m[v] = -__builtin_inff(); s[v] = 0.f; }

  for (int t = wv; t < NTILES; t += 8) {
    const int n0 = t * 16;
    const __bf16* bbase = wt + (size_t)(n0 + l15) * EMB + boff;
    __builtin_prefetch(bbase + 8 * 16 * EMB, 0, 1);   // next tile for this wave
    v8f acc = {};
#pragma unroll
    for (int c = 0; c < 4; ++c) {
      AB b; b.v = *(const v16bf*)(bbase + c * 32);
      acc = __builtin_amdgcn_wmma_f32_16x16x32_bf16(false, a[c].v, false, b.v,
                                                    (short)0, acc, false, false);
    }
#pragma unroll
    for (int v = 0; v < 8; ++v) {
      float x = acc[v];
      m[v] = fmaxf(m[v], x);
      s[v] += __expf(x);
    }
  }

#pragma unroll
  for (int v = 0; v < 8; ++v) { red_m[tid * 8 + v] = m[v]; red_s[tid * 8 + v] = s[v]; }
  __syncthreads();

  // ---- combine 8 waves x 16 lanes per row: plain fmax / add ----
  if (tid < 16) {
    const int M = tid;
    const int half = (M >= 8) ? 16 : 0;
    const int v = M & 7;
    float gm = -__builtin_inff(), gs = 0.f;
    for (int w = 0; w < 8; ++w)
#pragma unroll
      for (int l = 0; l < 16; ++l) {
        int t2 = w * 32 + half + l;
        gm = fmaxf(gm, red_m[t2 * 8 + v]);
        gs += red_s[t2 * 8 + v];
      }
    rowMax[M] = gm; rowInv[M] = 1.f / gs;
    rowW[M] = 1.f / (1.f + __expf(-gm));                       // sigmoid(top)
    float lp = (gm > 0.f) ? __logf(1.f + __expf(-gm))          // -log(sigmoid)
                          : (-gm + __logf(1.f + __expf(gm)));
    atomicAdd(lp_acc, lp);
  }
  __syncthreads();

  // ---- Pass 2: recompute logits; argmax by equality with rowmax; probs -> avg ----
  float gmr[8], li[8]; int ix[8];
#pragma unroll
  for (int v = 0; v < 8; ++v) {
    int M = v + (hi ? 8 : 0);
    gmr[v] = rowMax[M]; li[v] = rowInv[M]; ix[v] = 0x7FFFFFFF;
  }

  for (int t = wv; t < NTILES; t += 8) {
    const int n0 = t * 16;
    const __bf16* bbase = wt + (size_t)(n0 + l15) * EMB + boff;
    v8f acc = {};
#pragma unroll
    for (int c = 0; c < 4; ++c) {
      AB b; b.v = *(const v16bf*)(bbase + c * 32);
      acc = __builtin_amdgcn_wmma_f32_16x16x32_bf16(false, a[c].v, false, b.v,
                                                    (short)0, acc, false, false);
    }
    const int col = n0 + l15;
    float cs = 0.f;
#pragma unroll
    for (int v = 0; v < 8; ++v) {
      float x = acc[v];
      cs += __expf(x) * li[v];                          // p = exp(x)/sum(exp)
      ix[v] = (x >= gmr[v]) ? min(ix[v], col) : ix[v];  // branchless argmax
    }
    float tot = cs + xor16(cs);                 // lane pairs (n, n+16) share a column
    if (lane < 16) atomicAdd(&avg_acc[n0 + l15], tot);
  }

#pragma unroll
  for (int v = 0; v < 8; ++v) red_i[tid * 8 + v] = ix[v];
  __syncthreads();

  if (tid < 16) {
    const int M = tid;
    const int half = (M >= 8) ? 16 : 0;
    const int v = M & 7;
    int gi = 0x7FFFFFFF;
    for (int w = 0; w < 8; ++w)
#pragma unroll
      for (int l = 0; l < 16; ++l) gi = min(gi, red_i[(w * 32 + half + l) * 8 + v]);
    rowIdx[M] = gi;
  }
  __syncthreads();

  // ---- KV build: silu(w*pool[idx]) | silu(silu(hist@Wu)) ----
  {
    const int r   = tid >> 4;
    const int c0  = tid & 15;
    const int row = R0 + r;
    const float wgt = rowW[r];
    const float* prow = pool + (size_t)rowIdx[r] * GD;
    float* krow = KV + (size_t)row * EMB;
    for (int c = c0; c < GD; c += 16) krow[c] = silu(wgt * prow[c]);
    if (c0 < 8) {
      const float* hrow = hist_f32 + (size_t)row * EMB;
      float accu = 0.f;
      for (int e = 0; e < EMB; ++e) accu += hrow[e] * Wu[e * 8 + c0];
      krow[GD + c0] = silu(silu(accu));   // silu in branch, silu again outside
    }
  }
}

// One WG per batch element: silu(target@Wq) -> 4-head attention over s=40 -> @Wo
__global__ __launch_bounds__(128) void attn_kernel(
    const float* __restrict__ target, const float* __restrict__ Wq,
    const float* __restrict__ Kbuf, const float* __restrict__ Vbuf,
    const float* __restrict__ Wo, float* __restrict__ out)
{
  const int b = blockIdx.x, t = threadIdx.x;
  __shared__ float tv[128], qv[128], sc[160], hsum[4], ov[128];

  tv[t] = target[(size_t)b * 128 + t];
  __syncthreads();

  float acc = 0.f;
  for (int e = 0; e < 128; ++e) acc += tv[e] * Wq[e * 128 + t];
  qv[t] = silu(acc);
  __syncthreads();

  const float rs = 0.17677669529663687f;   // 1/sqrt(32)
  for (int idx = t; idx < 160; idx += 128) {
    int h = idx / 40, j = idx % 40;
    const float* kr = Kbuf + ((size_t)b * 40 + j) * 128 + h * 32;
    const float* qh = qv + h * 32;
    float s0 = 0.f;
    for (int d = 0; d < 32; ++d) s0 += qh[d] * kr[d];
    sc[h * 40 + j] = s0 * rs;
  }
  __syncthreads();

  if (t < 4) {
    float mx = -__builtin_inff();
    for (int j = 0; j < 40; ++j) mx = fmaxf(mx, sc[t * 40 + j]);
    float sm = 0.f;
    for (int j = 0; j < 40; ++j) { float p = __expf(sc[t * 40 + j] - mx); sc[t * 40 + j] = p; sm += p; }
    hsum[t] = sm;
  }
  __syncthreads();

  {
    int h = t >> 5, d = t & 31;
    float inv = 1.f / hsum[h];
    float o = 0.f;
    for (int j = 0; j < 40; ++j) o += sc[h * 40 + j] * Vbuf[((size_t)b * 40 + j) * 128 + h * 32 + d];
    ov[t] = o * inv;
  }
  __syncthreads();

  float fo = 0.f;
  for (int e = 0; e < 128; ++e) fo += ov[e] * Wo[e * 128 + t];
  out[(size_t)b * 128 + t] = fo;
}

__global__ void finalize_kernel(const float* __restrict__ avgK, const float* __restrict__ avgV,
                                const float* __restrict__ lp_acc, float* __restrict__ out_losses)
{
  __shared__ float red[256];
  float part = 0.f;
  for (int mi = threadIdx.x; mi < MPOOL; mi += 256) {
    float a = fmaxf(avgK[mi] * (1.0f / ROWS), 1e-9f);
    part += a * __logf(a * (float)MPOOL);
    float v = fmaxf(avgV[mi] * (1.0f / ROWS), 1e-9f);
    part += v * __logf(v * (float)MPOOL);
  }
  red[threadIdx.x] = part;
  __syncthreads();
  for (int st = 128; st > 0; st >>= 1) {
    if (threadIdx.x < st) red[threadIdx.x] += red[threadIdx.x + st];
    __syncthreads();
  }
  if (threadIdx.x == 0) {
    out_losses[0] = lp_acc[0] * (1.0f / ROWS);  // loss_peak
    out_losses[1] = red[0];                     // loss_balance
  }
}

extern "C" void kernel_launch(void* const* d_in, const int* in_sizes, int n_in,
                              void* d_out, int out_size, void* d_ws, size_t ws_size,
                              hipStream_t stream) {
  const float* target = (const float*)d_in[0];
  const float* hist   = (const float*)d_in[1];
  const float* Wq     = (const float*)d_in[2];
  const float* Wk     = (const float*)d_in[3];
  const float* Wv     = (const float*)d_in[4];
  const float* Wrk    = (const float*)d_in[5];
  const float* Wrv    = (const float*)d_in[6];
  const float* Kpool  = (const float*)d_in[7];
  const float* Vpool  = (const float*)d_in[8];
  const float* Wo     = (const float*)d_in[9];
  float* out = (float*)d_out;

  char* ws = (char*)d_ws;
  __bf16* histb = (__bf16*)(ws + 0);          // 2,621,440 B
  __bf16* wrkt  = (__bf16*)(ws + 2621440);    // 2,560,000 B
  __bf16* wrvt  = (__bf16*)(ws + 5181440);    // 2,560,000 B
  float*  Kbuf  = (float*)(ws + 7741440);     // 5,242,880 B
  float*  Vbuf  = (float*)(ws + 12984320);    // 5,242,880 B
  float*  avgK  = (float*)(ws + 18227200);    // 40,960 B
  float*  avgV  = (float*)(ws + 18268160);    // 40,960 B
  float*  lpacc = (float*)(ws + 18309120);    // 256 B

  hipMemsetAsync(ws + 18227200, 0, 40960 * 2 + 256, stream);

  cast_hist_kernel<<<(ROWS * EMB) / 256, 256, 0, stream>>>(hist, histb, ROWS * EMB);
  transpose_cast_kernel<<<(MPOOL * EMB) / 256, 256, 0, stream>>>(Wrk, wrkt);
  transpose_cast_kernel<<<(MPOOL * EMB) / 256, 256, 0, stream>>>(Wrv, wrvt);

  router_kernel<<<ROWS / 16, 256, 0, stream>>>(histb, wrkt, hist, Wk, Kpool, Kbuf, avgK, lpacc);
  router_kernel<<<ROWS / 16, 256, 0, stream>>>(histb, wrvt, hist, Wv, Vpool, Vbuf, avgV, lpacc);

  attn_kernel<<<NB, 128, 0, stream>>>(target, Wq, Kbuf, Vbuf, Wo, out);
  finalize_kernel<<<1, 256, 0, stream>>>(avgK, avgV, lpacc, out + NB * EMB);
}
